// CrossAttention_23716809409200
// MI455X (gfx1250) — matrix operand order repaired and checked by reading, not compiled, not run on hardware
//
#include <hip/hip_runtime.h>

typedef __attribute__((ext_vector_type(16))) _Float16 v16h;
typedef __attribute__((ext_vector_type(8)))  float    v8f;
typedef __attribute__((ext_vector_type(4)))  float    f4;

#define BS     1024
#define DM     768
#define KVL    64
#define NWAVE  8
#define DPW    96      // d-span per wave (3 chunks of K=32)
#define DTILE  128     // V rows staged per LDS tile
#define NTILE  6       // 768 / 128

__global__ __launch_bounds__(256)
void CrossAttention_23716809409200_kernel(const float* __restrict__ q,
                                          const float* __restrict__ K,
                                          const float* __restrict__ V,
                                          float* __restrict__ out)
{
    __shared__ float qlds[DM];              // 3 KB
    __shared__ float scp[NWAVE * KVL];      // 2 KB  per-wave score partials
    __shared__ float sfull[KVL];
    __shared__ float wexp[KVL];
    __shared__ float wts[KVL];
    __shared__ float ctxl[DTILE];
    __shared__ float vt[DTILE * KVL];       // 32 KB V tile

    const int b    = blockIdx.x;
    const int tid  = threadIdx.x;
    const int lane = tid & 31;
    const int w    = tid >> 5;

    const float* qb = q + (size_t)b * DM;
    const float* Kb = K + (size_t)b * DM * KVL;
    const float* Vb = V + (size_t)b * DM * KVL;
    float*       ob = out + (size_t)b * DM * KVL;

    // ---- stage q into LDS (coalesced) ----
    for (int i = tid; i < DM; i += 256) qlds[i] = qb[i];
    __syncthreads();

    // ---- Phase A: scores via v_wmma_f32_16x16x32_f16 ----
    // A (16x32 f16): lane m=lane&15 holds row m; lanes 0-15 hold K {0..7,16..23},
    // lanes 16-31 hold K {8..15,24..31}. Only row 0 = q, rows 1..15 = 0.
    // B (32x16 f16): lane n=lane&15 holds column n; lanes 0-15 hold K {0..7,16..23},
    // lanes 16-31 hold K {8..15,24..31} (same K order as A).
    const int   m       = lane & 15;
    const int   kgA     = lane >> 4;          // 0 or 1
    const int   n       = lane & 15;
    const int   kbB     = (lane >> 4) * 8;    // 0 or 8
    const float rowmask = (m == 0) ? 1.0f : 0.0f;   // branch-free row-0 select

    v8f acc[4] = {v8f{}, v8f{}, v8f{}, v8f{}};
    #pragma unroll
    for (int c = 0; c < 3; ++c) {
        const int d0 = w * DPW + c * 32;
        // Needed q K-values are two contiguous 8-float runs -> 4x ds_load_b128.
        const f4* qlo = (const f4*)&qlds[d0 + kgA * 8];        // K = kg*8 + 0..7
        const f4* qhi = (const f4*)&qlds[d0 + 16 + kgA * 8];   // K = 16 + kg*8 + 0..7
        f4 x0 = qlo[0] * rowmask;
        f4 x1 = qlo[1] * rowmask;
        f4 x2 = qhi[0] * rowmask;
        f4 x3 = qhi[1] * rowmask;
        v16h a;
        a[0]  = (_Float16)x0.x;  a[1]  = (_Float16)x0.y;
        a[2]  = (_Float16)x0.z;  a[3]  = (_Float16)x0.w;
        a[4]  = (_Float16)x1.x;  a[5]  = (_Float16)x1.y;
        a[6]  = (_Float16)x1.z;  a[7]  = (_Float16)x1.w;
        a[8]  = (_Float16)x2.x;  a[9]  = (_Float16)x2.y;
        a[10] = (_Float16)x2.z;  a[11] = (_Float16)x2.w;
        a[12] = (_Float16)x3.x;  a[13] = (_Float16)x3.y;
        a[14] = (_Float16)x3.z;  a[15] = (_Float16)x3.w;

        #pragma unroll
        for (int t = 0; t < 4; ++t) {
            const int kv0 = t * 16;
            v16h bm;
            #pragma unroll
            for (int i = 0; i < 16; ++i) {
                // lanes 0-15 / 16-31 each hit a 64B-aligned 16-float row -> coalesced
                bm[i] = (_Float16)Kb[(d0 + kbB + (i & 7) + ((i >> 3) << 4)) * KVL + kv0 + n];
            }
            acc[t] = __builtin_amdgcn_wmma_f32_16x16x32_f16(
                false, a, false, bm, (short)0, acc[t], false, false);
        }
    }
    // Row M=0 of each C tile: lanes 0-15, VGPR0 (element 0).
    if (lane < 16) {
        scp[w * KVL +  0 + lane] = acc[0][0];
        scp[w * KVL + 16 + lane] = acc[1][0];
        scp[w * KVL + 32 + lane] = acc[2][0];
        scp[w * KVL + 48 + lane] = acc[3][0];
    }
    __syncthreads();

    // ---- Phase B: reduce partials + softmax over 64 kv ----
    if (tid < KVL) {
        float s = 0.f;
        #pragma unroll
        for (int ww = 0; ww < NWAVE; ++ww) s += scp[ww * KVL + tid];
        sfull[tid] = s;
    }
    __syncthreads();
    if (tid < KVL) {
        float mx = sfull[0];
        for (int k = 1; k < KVL; ++k) mx = fmaxf(mx, sfull[k]);
        wexp[tid] = __expf(sfull[tid] - mx);
    }
    __syncthreads();
    if (tid < KVL) {
        float sum = 0.f;
        for (int k = 0; k < KVL; ++k) sum += wexp[k];
        wts[tid] = wexp[tid] / sum;
    }
    __syncthreads();

    // ---- Phase C: per-tile V -> LDS (async), ctx, out = ctx * V ----
    for (int T = 0; T < NTILE; ++T) {
        const int d0 = T * DTILE;

        // async copy 32 KB V tile into LDS: 2048 float4, 8 per thread, coalesced
        #pragma unroll
        for (int j = 0; j < 8; ++j) {
            const int vi = tid + j * 256;                      // float4 index
            unsigned lds_off = (unsigned)(uintptr_t)(&vt[vi * 4]);
            const float* gp = Vb + d0 * KVL + vi * 4;
            asm volatile("global_load_async_to_lds_b128 %0, %1, off"
                         :: "v"(lds_off), "v"(gp) : "memory");
        }
        asm volatile("s_wait_asynccnt 0" ::: "memory");
        __syncthreads();

        // ctx[r] = sum_k wts[k] * V[r][k]; two threads per row, combine via shfl
        {
            const int r = tid >> 1;
            const int h = tid & 1;
            const f4* vr4 = (const f4*)&vt[r * KVL + h * 32];
            const f4* wt4 = (const f4*)&wts[h * 32];
            float p = 0.f;
            #pragma unroll
            for (int k4 = 0; k4 < 8; ++k4) {
                f4 a4 = vr4[k4];
                f4 w4 = wt4[k4];
                p += a4.x * w4.x + a4.y * w4.y + a4.z * w4.z + a4.w * w4.w;
            }
            p += __shfl_xor(p, 1);
            if (h == 0) ctxl[r] = p;
        }
        __syncthreads();

        // out[d][k] = ctx[d] * V[d][k], fully coalesced float4 stores
        const f4* vt4 = (const f4*)vt;
        f4* o4 = (f4*)(ob + d0 * KVL);
        #pragma unroll
        for (int j = 0; j < 8; ++j) {
            const int vi  = tid + j * 256;
            const int row = vi >> 4;            // 16 float4 per 64-wide row
            f4 v4 = vt4[vi];
            o4[vi] = v4 * ctxl[row];
        }
        __syncthreads();   // protect vt before next tile's async fill
    }
}

extern "C" void kernel_launch(void* const* d_in, const int* in_sizes, int n_in,
                              void* d_out, int out_size, void* d_ws, size_t ws_size,
                              hipStream_t stream) {
    (void)in_sizes; (void)n_in; (void)out_size; (void)d_ws; (void)ws_size;
    const float* q = (const float*)d_in[0];
    const float* K = (const float*)d_in[1];
    const float* V = (const float*)d_in[2];
    float* out = (float*)d_out;
    hipLaunchKernelGGL(CrossAttention_23716809409200_kernel,
                       dim3(BS), dim3(256), 0, stream, q, K, V, out);
}
